// GCNlayer_24592982737596
// MI455X (gfx1250) — matrix operand order repaired and checked by reading, not compiled, not run on hardware
//
#include <hip/hip_runtime.h>
#include <hip/hip_bf16.h>
#include <stdint.h>

#define B_     8
#define C_     128
#define HW_    65536
#define K_     64
#define NPIX   (B_*HW_)
#define BN_EPS 1e-5f

typedef __attribute__((ext_vector_type(16))) __bf16   v16bf;
typedef __attribute__((ext_vector_type(8)))  float    v8f;
typedef __attribute__((ext_vector_type(8)))  uint32_t v8u;

union U8 { v8u v; uint4 q[2]; };

// ---- workspace layout (4-byte elements) ----
#define WT_HI_OFF   0          // 8192 u32  : bf16-hi pairs of weight^T
#define WT_LO_OFF   8192       // 8192 u32  : bf16-lo pairs of weight^T
#define SUMS_OFF    16384      // 65536 f32 : per (b,k,d) cluster sums
#define COUNTS_OFF  81920      // 512  u32  : per (b,k) counts
#define BNSUM_OFF   82432      // 128  f32
#define BNSQ_OFF    82560      // 128  f32
#define ADJM_OFF    82688      // 65536 f32 : adj_means (b,k,d)
#define ZERO_BEG    16384
#define ZERO_CNT    66304      // sums + counts + bnsum + bnsq

static __device__ __forceinline__ uint16_t f2bf(float f) {
  uint32_t u = __builtin_bit_cast(uint32_t, f);
  u += 0x7FFFu + ((u >> 16) & 1u);          // round-to-nearest-even truncation
  return (uint16_t)(u >> 16);
}
static __device__ __forceinline__ float bf2f(uint16_t h) {
  return __builtin_bit_cast(float, (uint32_t)h << 16);
}

// ---------------------------------------------------------------------------
__global__ void k_zero(float* __restrict__ ws) {
  int i = blockIdx.x * 256 + threadIdx.x;
  if (i < ZERO_CNT) ws[ZERO_BEG + i] = 0.0f;
}

// weight (Cin x Cout) -> transposed bf16 hi/lo pair tables: wt*[n*64 + c2]
// packs {bf16(w[2*c2][n]) , bf16(w[2*c2+1][n])} = one WMMA B-fragment dword.
__global__ void k_splitw(const float* __restrict__ w,
                         uint32_t* __restrict__ wtHi, uint32_t* __restrict__ wtLo) {
  int i = blockIdx.x * 256 + threadIdx.x;       // i = n*64 + c2
  if (i >= C_ * 64) return;
  int n = i >> 6, c2 = i & 63;
  float v0 = w[(2*c2    ) * C_ + n];
  float v1 = w[(2*c2 + 1) * C_ + n];
  uint16_t h0 = f2bf(v0), h1 = f2bf(v1);
  uint16_t l0 = f2bf(v0 - bf2f(h0)), l1 = f2bf(v1 - bf2f(h1));
  wtHi[i] = (uint32_t)h0 | ((uint32_t)h1 << 16);
  wtLo[i] = (uint32_t)l0 | ((uint32_t)l1 << 16);
}

// per-(b,k) pixel histogram: LDS accumulate, then flush
__global__ void k_hist(const int* __restrict__ index, uint32_t* __restrict__ counts) {
  __shared__ uint32_t h[K_];
  int tid = threadIdx.x;
  if (tid < K_) h[tid] = 0;
  __syncthreads();
  int p = blockIdx.x * 256 + tid;
  atomicAdd(&h[index[p] & 63], 1u);
  __syncthreads();
  int b = (blockIdx.x * 256) >> 16;
  if (tid < K_ && h[tid]) atomicAdd(&counts[b * K_ + tid], h[tid]);
}

// ---------------------------------------------------------------------------
// GEMM: xw = x @ weight  (split-bf16: hi*hi + hi*lo + lo*hi per K-chunk of 32)
// block = 256 thr = 8 waves; tile = 64 pixels x 128 out-channels.
// Each wave: one 16-col strip, 4 row-subtiles -> 48 v_wmma per wave.
// Also scatter-adds xw into cluster sums (the one-hot einsum).
__global__ void __launch_bounds__(256)
k_gemm(const float* __restrict__ x, const int* __restrict__ index,
       const uint32_t* __restrict__ wtHi, const uint32_t* __restrict__ wtLo,
       float* __restrict__ xw, float* __restrict__ sums) {
  __shared__ uint32_t aHi[4096];   // [m][c2 ^ ((m&15)<<2)] packed bf16-hi pairs
  __shared__ uint32_t aLo[4096];
  __shared__ int kIdx[64];

  int tid  = threadIdx.x;
  int pix0 = blockIdx.x << 6;            // 64 pixels per block (never crosses batch)
  int b    = pix0 >> 16;
  int hw0  = pix0 & (HW_ - 1);

  // cooperative A load + split: lanes walk consecutive pixels (coalesced 128B)
  for (int i = tid; i < 4096; i += 256) {
    int c2 = i >> 6, m = i & 63;
    const float* xp = x + (((b * C_ + 2 * c2) << 16) + hw0 + m);
    float v0 = xp[0];
    float v1 = xp[1 << 16];
    uint16_t h0 = f2bf(v0), h1 = f2bf(v1);
    uint16_t l0 = f2bf(v0 - bf2f(h0)), l1 = f2bf(v1 - bf2f(h1));
    int s = (m << 6) + (c2 ^ ((m & 15) << 2));      // keeps 4-groups contiguous
    aHi[s] = (uint32_t)h0 | ((uint32_t)h1 << 16);
    aLo[s] = (uint32_t)l0 | ((uint32_t)l1 << 16);
  }
  if (tid < 64) kIdx[tid] = index[pix0 + tid] & 63;
  __syncthreads();

  int lane = tid & 31;
  int wave = tid >> 5;
  int half = lane >> 4;
  int nrow = lane & 15;
  int n    = (wave << 4) + nrow;   // output channel (B/C column)
  int sw   = nrow << 2;            // lane's A swizzle (m&15 == nrow for all subtiles)

  v8f acc[4] = {v8f{}, v8f{}, v8f{}, v8f{}};
  const uint4* wH = (const uint4*)(wtHi + (n << 6));
  const uint4* wL = (const uint4*)(wtLo + (n << 6));

  #pragma unroll
  for (int kc = 0; kc < 4; ++kc) {       // K chunks of 32 input channels
    int k0h = kc << 4;                   // k0/2
    // B fragments: K = 16*half + 2*v (+s)  ->  8 consecutive pairs in row n
    int cb = (kc << 2) + (half << 1);    // uint4 index of pair (k0h + 8*half)
    U8 ubh, ubl;
    ubh.q[0] = wH[cb]; ubh.q[1] = wH[cb + 1];
    ubl.q[0] = wL[cb]; ubl.q[1] = wL[cb + 1];
    v16bf bh = __builtin_bit_cast(v16bf, ubh.v);
    v16bf bl = __builtin_bit_cast(v16bf, ubl.v);

    int c2lo = k0h + (half << 3);        // A pair base: v=0..3 ; +4 for v=4..7
    int j0 = c2lo ^ sw;
    int j1 = (c2lo + 4) ^ sw;
    #pragma unroll
    for (int mt = 0; mt < 4; ++mt) {     // 4 row-subtiles of 16 pixels
      int arow = ((mt << 4) + nrow) << 6;
      U8 uah, ual;
      uah.q[0] = *(const uint4*)&aHi[arow + j0];
      uah.q[1] = *(const uint4*)&aHi[arow + j1];
      ual.q[0] = *(const uint4*)&aLo[arow + j0];
      ual.q[1] = *(const uint4*)&aLo[arow + j1];
      v16bf ah = __builtin_bit_cast(v16bf, uah.v);
      v16bf al = __builtin_bit_cast(v16bf, ual.v);
      acc[mt] = __builtin_amdgcn_wmma_f32_16x16x32_bf16(false, ah, false, bh, (short)0, acc[mt], false, false);
      acc[mt] = __builtin_amdgcn_wmma_f32_16x16x32_bf16(false, ah, false, bl, (short)0, acc[mt], false, false);
      acc[mt] = __builtin_amdgcn_wmma_f32_16x16x32_bf16(false, al, false, bh, (short)0, acc[mt], false, false);
    }
  }

  // C/D layout: VGPR r -> pixel m = 16*mt + 8*half + r  (contiguous per subtile)
  float* plane = xw + ((b * C_ + n) << 16) + hw0;
  #pragma unroll
  for (int mt = 0; mt < 4; ++mt) {
    float* op = plane + (mt << 4) + (half << 3);
    *(float4*)(op)     = make_float4(acc[mt][0], acc[mt][1], acc[mt][2], acc[mt][3]);
    *(float4*)(op + 4) = make_float4(acc[mt][4], acc[mt][5], acc[mt][6], acc[mt][7]);
    #pragma unroll
    for (int r = 0; r < 8; ++r) {
      int mm = (mt << 4) + (half << 3) + r;
      atomicAdd(&sums[((b * K_ + kIdx[mm]) << 7) + n], acc[mt][r]);
    }
  }
}

// ---------------------------------------------------------------------------
// Per-batch graph math. quad = D (W W^T) D^T = ||Y_p - Y_q||^2 with Y = means@W.
__global__ void k_graph(const float* __restrict__ sums, const uint32_t* __restrict__ counts,
                        const float* __restrict__ Wm, const float* __restrict__ adj_mask,
                        float* __restrict__ adjm) {
  __shared__ float meansS[K_ * C_];   // 32 KB
  __shared__ float Ys[K_ * C_];       // 32 KB
  __shared__ float adjS[K_ * K_];     // 16 KB
  int b = blockIdx.x, tid = threadIdx.x;

  for (int i = tid; i < K_ * C_; i += 256) {
    int p = i >> 7;
    float cnt   = (float)counts[b * K_ + p];
    float denom = cnt + (cnt == 0.0f ? 1.0f : 0.0f);
    meansS[i] = sums[(b * K_ << 7) + i] / denom;
  }
  __syncthreads();
  for (int i = tid; i < K_ * C_; i += 256) {
    int p = i >> 7, e = i & 127;
    float s = 0.0f;
    for (int d = 0; d < C_; ++d) s += meansS[(p << 7) + d] * Wm[(d << 7) + e];
    Ys[i] = s;
  }
  __syncthreads();
  for (int i = tid; i < K_ * K_; i += 256) {
    int p = i >> 6, q = i & 63;
    float a = 0.0f;
    if (p != q) {
      float s = 0.0f;
      for (int e = 0; e < C_; ++e) {
        float dl = Ys[(p << 7) + e] - Ys[(q << 7) + e];
        s += dl * dl;
      }
      a = __expf(-s) * adj_mask[i];   // i = p*64+q
    }
    adjS[i] = a;
  }
  __syncthreads();
  for (int i = tid; i < K_ * C_; i += 256) {
    int p = i >> 7, d = i & 127;
    float s = 0.0f;
    for (int q = 0; q < K_; ++q) s += adjS[(p << 6) + q] * meansS[(q << 7) + d];
    adjm[(b * K_ << 7) + i] = s;
  }
}

// ---------------------------------------------------------------------------
// features = xw + gather(adj_means) in place in d_out; BN partial sums.
__global__ void k_feat(const int* __restrict__ index, const float* __restrict__ adjm,
                       float* __restrict__ feat,
                       float* __restrict__ bnsum, float* __restrict__ bnsq) {
  __shared__ float am[K_];
  __shared__ float red[256], red2[256];
  int g = blockIdx.x;
  int chunk = g & 7, d = (g >> 3) & 127, b = g >> 10;
  int tid = threadIdx.x;
  if (tid < K_) am[tid] = adjm[((b * K_ + tid) << 7) + d];
  __syncthreads();

  float*     fp = feat  + ((b * C_ + d) << 16) + (chunk << 13);
  const int* ip = index + (b << 16)            + (chunk << 13);
  float s = 0.0f, s2 = 0.0f;
  #pragma unroll
  for (int it = 0; it < 8; ++it) {
    int off = ((it << 8) + tid) << 2;
    float4 xv = *(const float4*)(fp + off);
    int4   iv = *(const int4*)(ip + off);
    float4 f;
    f.x = xv.x + am[iv.x & 63];
    f.y = xv.y + am[iv.y & 63];
    f.z = xv.z + am[iv.z & 63];
    f.w = xv.w + am[iv.w & 63];
    *(float4*)(fp + off) = f;
    s  += f.x + f.y + f.z + f.w;
    s2 += f.x * f.x + f.y * f.y + f.z * f.z + f.w * f.w;
  }
  red[tid] = s; red2[tid] = s2;
  __syncthreads();
  for (int st = 128; st > 0; st >>= 1) {
    if (tid < st) { red[tid] += red[tid + st]; red2[tid] += red2[tid + st]; }
    __syncthreads();
  }
  if (tid == 0) {
    atomicAdd(&bnsum[d], red[0]);
    atomicAdd(&bnsq[d],  red2[0]);
  }
}

// ---------------------------------------------------------------------------
__global__ void k_bn(float* __restrict__ out,
                     const float* __restrict__ bnsum, const float* __restrict__ bnsq,
                     const float* __restrict__ gamma, const float* __restrict__ beta) {
  int g = blockIdx.x;
  int chunk = g & 7, d = (g >> 3) & 127, b = g >> 10;
  int tid = threadIdx.x;
  const float invN = 1.0f / (float)(B_ * HW_);
  float mu  = bnsum[d] * invN;
  float var = bnsq[d] * invN - mu * mu;
  float sc  = gamma[d] * rsqrtf(var + BN_EPS);
  float sh  = beta[d] - mu * sc;
  float* fp = out + ((b * C_ + d) << 16) + (chunk << 13);
  #pragma unroll
  for (int it = 0; it < 8; ++it) {
    int off = ((it << 8) + tid) << 2;
    float4 v = *(float4*)(fp + off);
    v.x = v.x * sc + sh; v.y = v.y * sc + sh;
    v.z = v.z * sc + sh; v.w = v.w * sc + sh;
    *(float4*)(fp + off) = v;
  }
}

// ---------------------------------------------------------------------------
extern "C" void kernel_launch(void* const* d_in, const int* in_sizes, int n_in,
                              void* d_out, int out_size, void* d_ws, size_t ws_size,
                              hipStream_t stream) {
  (void)in_sizes; (void)n_in; (void)out_size; (void)ws_size;
  const float* x      = (const float*)d_in[0];
  const int*   index  = (const int*)d_in[1];
  const float* weight = (const float*)d_in[2];
  const float* Wm     = (const float*)d_in[3];
  const float* adjmsk = (const float*)d_in[4];
  const float* gamma  = (const float*)d_in[5];
  const float* beta   = (const float*)d_in[6];
  float* out = (float*)d_out;
  float* ws  = (float*)d_ws;

  uint32_t* wtHi   = (uint32_t*)(ws + WT_HI_OFF);
  uint32_t* wtLo   = (uint32_t*)(ws + WT_LO_OFF);
  float*    sums   = ws + SUMS_OFF;
  uint32_t* counts = (uint32_t*)(ws + COUNTS_OFF);
  float*    bnsum  = ws + BNSUM_OFF;
  float*    bnsq   = ws + BNSQ_OFF;
  float*    adjm   = ws + ADJM_OFF;

  k_zero  <<<(ZERO_CNT + 255) / 256, 256, 0, stream>>>(ws);
  k_splitw<<<(C_ * 64 + 255) / 256,  256, 0, stream>>>(weight, wtHi, wtLo);
  k_hist  <<<NPIX / 256,             256, 0, stream>>>(index, counts);
  k_gemm  <<<NPIX / 64,              256, 0, stream>>>(x, index, wtHi, wtLo, out, sums);
  k_graph <<<B_,                     256, 0, stream>>>(sums, counts, Wm, adjmsk, adjm);
  k_feat  <<<B_ * C_ * 8,            256, 0, stream>>>(index, adjm, out, bnsum, bnsq);
  k_bn    <<<B_ * C_ * 8,            256, 0, stream>>>(out, bnsum, bnsq, gamma, beta);
}
